// Kernel_12275016531969
// MI455X (gfx1250) — matrix operand order repaired and checked by reading, not compiled, not run on hardware
//
#include <hip/hip_runtime.h>
#include <math.h>

// ---------------- types ----------------
typedef _Float16 h8   __attribute__((ext_vector_type(8)));
typedef _Float16 v16h __attribute__((ext_vector_type(16)));
typedef float    v8f  __attribute__((ext_vector_type(8)));
typedef float    f4v  __attribute__((ext_vector_type(4)));

// ---------------- problem constants (from reference) ----------------
#define NOUT      144
#define NPATHCOL  4864
#define HIDDEN    100
#define KPAD      128        // K padded for 4 x k32 WMMA steps
#define HSTRIDE   136        // LDS halves stride for h tile (bank-spread)
#define SCSTRIDE  2576       // LDS floats stride for coeff tile (not mult of 64)
#define NTILES    304        // 4864/16 N tiles

// 19 paths in coeff order: (lo,li) blocks, lf inner
constexpr int LOA_C[19] = {0,0,0,1,1,1,1,1,1,1,2,2,2,2,2,2,2,2,2};
constexpr int LIA_C[19] = {0,1,2,0,1,1,1,2,2,2,0,1,1,1,2,2,2,2,2};
constexpr int LFA_C[19] = {0,1,2,1,0,1,2,1,2,3,2,1,2,3,0,1,2,3,4};
constexpr int CGB_C[19] = {0,1,10,35,44,53,80,125,170,245,350,375,420,495,600,625,700,825,1000}; // CG flat offsets (total 1225)
constexpr int CYB_C[19] = {0,1,4,9,12,21,30,39,54,69,84,89,104,119,134,159,184,209,234};        // CY per-point offsets (total 259)
// per (lo,li) block
constexpr int NLF3[9] = {1,1,1,1,3,3,1,3,5};
constexpr int PST3[9] = {0,1,2,3,4,7,10,11,14};
constexpr int CB3 [9] = {0,256,512,768,1024,1792,2560,2816,3584};
constexpr int RB3 [3] = {0,16,64};

// ---------------- LDS layout (bytes) ----------------
constexpr int OFF_SY    = 0;                    // 16 x 32 f32 (25 Ys + radius + iszero)
constexpr int OFF_SH    = 2048;                 // 16 x 136 f16
constexpr int OFF_SCY   = OFF_SH    + 4352;     // 16 x 260 f32
constexpr int OFF_SNORM = OFF_SCY   + 16640;    // 18 f32 (padded 128B)
constexpr int OFF_SCG   = OFF_SNORM + 128;      // 1225 f32 (padded 4928B)
constexpr int OFF_SC    = OFF_SCG   + 4928;     // 16 x 2576 f32 coeff tile
constexpr int SMEM_BYTES = OFF_SC + 16*SCSTRIDE*4;   // 192,960 B

// =================================================================
// Init kernel 1: real Clebsch-Gordan tensors (reference math, fp64)
// =================================================================
__device__ inline double dfact(int n){ double f=1.0; for(int i=2;i<=n;i++) f*=(double)i; return f; }

__device__ inline void buildQ(int l, double* qr, double* qi){
  int d=2*l+1;
  for (int a=0;a<d*d;a++){ qr[a]=0.0; qi[a]=0.0; }
  qr[l*d+l]=1.0;
  double s2 = 1.0/sqrt(2.0);
  for (int m=1;m<=l;m++){
    double sg = (m&1)? -1.0:1.0;
    qr[(l+m)*d+(l+m)] = sg*s2;
    qr[(l+m)*d+(l-m)] = s2;
    qi[(l-m)*d+(l-m)] = s2;
    qi[(l-m)*d+(l+m)] = -sg*s2;
  }
}

__global__ void init_cg_kernel(float* __restrict__ cg, float* __restrict__ normTab){
  const int pa = threadIdx.x;
  if (pa < 19) {
    const int j1=LOA_C[pa], j2=LIA_C[pa], j3=LFA_C[pa];
    const int d1=2*j1+1, d2=2*j2+1, d3=2*j3+1;
    double C[225], W[225];
    for (int q=0;q<d1*d2*d3;q++) C[q]=0.0;
    double pref0 = sqrt((double)(2*j3+1)*dfact(j1+j2-j3)*dfact(j1-j2+j3)*dfact(-j1+j2+j3)/dfact(j1+j2+j3+1));
    for (int m1=-j1;m1<=j1;m1++)
      for (int m2=-j2;m2<=j2;m2++){
        int m3=m1+m2; if (m3<-j3||m3>j3) continue;
        double pref = pref0*sqrt(dfact(j3+m3)*dfact(j3-m3)*dfact(j1-m1)*dfact(j1+m1)*dfact(j2-m2)*dfact(j2+m2));
        int klo = 0; if (j2-j3-m1>klo) klo=j2-j3-m1; if (j1+m2-j3>klo) klo=j1+m2-j3;
        int khi = j1+j2-j3; if (j1-m1<khi) khi=j1-m1; if (j2+m2<khi) khi=j2+m2;
        double s=0.0;
        for (int k=klo;k<=khi;k++){
          double den = dfact(k)*dfact(j1+j2-j3-k)*dfact(j1-m1-k)*dfact(j2+m2-k)*dfact(j3-j2+m1+k)*dfact(j3-j1-m2+k);
          s += ((k&1)? -1.0:1.0)/den;
        }
        C[((j1+m1)*d2+(j2+m2))*d3 + (j3+m3)] = pref*s;
      }
    // Wigner 3j from CG
    double is3 = 1.0/sqrt((double)(2*j3+1));
    for (int a=0;a<d1;a++) for (int b=0;b<d2;b++)
      for (int m3=-j3;m3<=j3;m3++){
        int e = j1 - j2 + m3;
        double sgn = (e&1)? -1.0 : 1.0;
        W[(a*d2+b)*d3 + (j3+m3)] = sgn*is3*C[(a*d2+b)*d3 + (j3-m3)];
      }
    // change to real basis:  c[a,b,c] = sum Q1[a,m] Q2[b,n] Q3[c,k] W[m,n,k]
    double q1r[25],q1i[25],q2r[25],q2i[25],q3r[81],q3i[81];
    buildQ(j1,q1r,q1i); buildQ(j2,q2r,q2i); buildQ(j3,q3r,q3i);
    double cr[225], ci[225], nr=0.0, ni=0.0;
    for (int a=0;a<d1;a++) for (int b=0;b<d2;b++) for (int c=0;c<d3;c++){
      double sr=0.0, si=0.0;
      for (int m=0;m<d1;m++){
        double ar=q1r[a*d1+m], ai=q1i[a*d1+m];
        if (ar==0.0 && ai==0.0) continue;
        for (int n=0;n<d2;n++){
          double br=q2r[b*d2+n], bi=q2i[b*d2+n];
          if (br==0.0 && bi==0.0) continue;
          double abr = ar*br - ai*bi, abi = ar*bi + ai*br;
          for (int k=0;k<d3;k++){
            double w = W[(m*d2+n)*d3+k];
            if (w==0.0) continue;
            double zr=q3r[c*d3+k], zi=q3i[c*d3+k];
            sr += (abr*zr - abi*zi)*w;
            si += (abr*zi + abi*zr)*w;
          }
        }
      }
      int idx=(a*d2+b)*d3+c;
      cr[idx]=sr; ci[idx]=si; nr+=sr*sr; ni+=si*si;
    }
    const double* sel = (nr>=ni)? cr : ci;
    double inv = 1.0/sqrt((nr>=ni)? nr : ni);
    for (int q=0;q<d1*d2*d3;q++) cg[CGB_C[pa]+q] = (float)(sel[q]*inv);
  }
  if (pa == 19) {
    const double PI_D = 3.14159265358979323846;
    const int num[3] = {48,112,144};
    for (int lo=0;lo<3;lo++) for (int li=0;li<3;li++){
      double lm = sqrt((double)(2*li+1)) * sqrt(4.0*PI_D);
      normTab[(lo*3+li)*2+0] = (float)(lm/sqrt((double)num[lo]));
      normTab[(lo*3+li)*2+1] = (float)(lm/4.0);
    }
  }
}

// =================================================================
// Init kernel 2: W2 (f32, 100x4864) -> f16 B-fragments, contiguous k order
// per tile t, kstep s, lane L, elem e:  N = t*16+(L&15), K = 32s+(L&16)+e
// =================================================================
__global__ void swizzle_w2(const float* __restrict__ W2, _Float16* __restrict__ w2h){
  int idx = blockIdx.x*256 + threadIdx.x;
  if (idx >= NTILES*2048) return;
  int e = idx & 15, L = (idx>>4)&31, s = (idx>>9)&3, t = idx>>11;
  int col = t*16 + (L & 15);
  int K   = s*32 + (L & 16) + e;
  float v = (K < HIDDEN) ? W2[(size_t)K*NPATHCOL + col] : 0.f;
  w2h[idx] = (_Float16)v;
}

// =================================================================
// Fused main kernel helpers
// =================================================================
__device__ inline v16h loadA(const _Float16* sH, int m, int kbase, int s){
  const _Float16* hp = sH + m*HSTRIDE + 32*s + kbase;
  h8 lo = *(const h8*)hp;
  h8 hi = *(const h8*)(hp + 8);
  return __builtin_shufflevector(lo,hi,0,1,2,3,4,5,6,7,8,9,10,11,12,13,14,15);
}

template<int PA>
__device__ inline void computeCY(float* sCY, const float* sY, const float* sCG, int tid){
  constexpr int di = 2*LOA_C[PA]+1;
  constexpr int dj = 2*LIA_C[PA]+1;
  constexpr int dk = 2*LFA_C[PA]+1;
  constexpr int yo = LFA_C[PA]*LFA_C[PA];
  for (int idx = tid; idx < 16*di*dj; idx += 256) {
    int p = idx / (di*dj);
    int e = idx - p*(di*dj);
    float acc = 0.f;
#pragma unroll
    for (int k=0;k<dk;k++)
      acc += sCG[CGB_C[PA] + e*dk + k] * sY[p*32 + yo + k];
    sCY[p*260 + CYB_C[PA] + e] = acc;
  }
}

template<int C0, int NT>
__device__ inline void gemmHalf(float* sC, const _Float16* __restrict__ w2h,
                                v16h a0, v16h a1, v16h a2, v16h a3, int tid){
  const int wv   = tid >> 5;
  const int lane = tid & 31;
  const int mb   = (lane & 16) >> 1;   // 0 or 8 (C-fragment M base)
  const v16h* w2v = (const v16h*)w2h;
  for (int t = wv; t < NT; t += 8) {
    const int tg = (C0/16) + t;
    v8f acc = {};
    v16h b0 = w2v[(size_t)(tg*4+0)*32 + lane];
    acc = __builtin_amdgcn_wmma_f32_16x16x32_f16(false, a0, false, b0, (short)0, acc, false, false);
    v16h b1 = w2v[(size_t)(tg*4+1)*32 + lane];
    acc = __builtin_amdgcn_wmma_f32_16x16x32_f16(false, a1, false, b1, (short)0, acc, false, false);
    v16h b2 = w2v[(size_t)(tg*4+2)*32 + lane];
    acc = __builtin_amdgcn_wmma_f32_16x16x32_f16(false, a2, false, b2, (short)0, acc, false, false);
    v16h b3 = w2v[(size_t)(tg*4+3)*32 + lane];
    acc = __builtin_amdgcn_wmma_f32_16x16x32_f16(false, a3, false, b3, (short)0, acc, false, false);
    float* dst = sC + (t*16 + (lane & 15));
#pragma unroll
    for (int rr = 0; rr < 8; rr++)
      dst[(size_t)(mb + rr) * SCSTRIDE] = acc[rr];   // C-frag: VGPR r -> M=r(+8)
  }
}

template<int LO, int LI, int C0>
__device__ inline void expandBlock(const float* sC, const float* sCY, const float* sY,
                                   const float* sNorm, float* __restrict__ out,
                                   int z0, int npts, int tid){
  constexpr int blk = LO*3+LI;
  constexpr int nlf = NLF3[blk];
  constexpr int ps  = PST3[blk];
  constexpr int cb  = CB3[blk];
  constexpr int rb  = RB3[LO];
  constexpr int cbc = RB3[LI];
  constexpr int Wd  = 2*LO+1, Wj = 2*LI+1;
  constexpr int R   = 16*Wd,  C  = 16*Wj, C4 = C/4;
  for (int idx = tid; idx < 16*R*C4; idx += 256) {
    int p    = idx / (R*C4);
    int rem  = idx - p*(R*C4);
    int rowL = rem / C4;
    int colL = (rem - rowL*C4)*4;
    int u = rowL / Wd, i = rowL - u*Wd;
    int isz = (sY[p*32+26] != 0.f) ? 1 : 0;
    float nrm = sNorm[blk*2 + isz];
    const float* cyp = sCY + p*260;
    const float* cp0 = sC + (size_t)p*SCSTRIDE + (cb - C0);
    f4v val;
#pragma unroll
    for (int q=0;q<4;q++){
      int cl = colL + q;
      int v = cl / Wj, j = cl - v*Wj;
      float acc = 0.f;
      const float* cp = cp0 + (u*16+v)*nlf;
#pragma unroll
      for (int f=0; f<nlf; f++)
        acc += cyp[CYB_C[ps+f] + i*Wj + j] * cp[f];
      val[q] = acc*nrm;
    }
    if (z0 + p < npts) {
      size_t o = (size_t)(z0+p)*(NOUT*NOUT) + (size_t)(rb+rowL)*NOUT + (cbc+colL);
      __builtin_nontemporal_store(val, (f4v*)(out + o));
    }
  }
}

// =================================================================
// Fused kernel: 1 workgroup = 16 points; SH+h+CY -> WMMA GEMM (two
// column halves, coeff in LDS) -> CG expansion -> coalesced NT stores
// =================================================================
__global__ void __launch_bounds__(256)
fused_kernel(const float* __restrict__ r, const float* __restrict__ W1,
             const float* __restrict__ b1, const float* __restrict__ cgG,
             const float* __restrict__ normG, const _Float16* __restrict__ w2h,
             float* __restrict__ out, int npts){
  extern __shared__ char smem[];
  float*    sY    = (float*)   (smem + OFF_SY);
  _Float16* sH    = (_Float16*)(smem + OFF_SH);
  float*    sCY   = (float*)   (smem + OFF_SCY);
  float*    sNorm = (float*)   (smem + OFF_SNORM);
  float*    sCG   = (float*)   (smem + OFF_SCG);
  float*    sC    = (float*)   (smem + OFF_SC);

  const int tid = threadIdx.x;
  const int z0  = blockIdx.x * 16;

  // stage CG + norm tables
  for (int i2 = tid; i2 < 1225; i2 += 256) sCG[i2] = cgG[i2];
  if (tid < 18) sNorm[tid] = normG[tid];

  // ---- phase A1: spherical harmonics (threads 0..15, one point each) ----
  if (tid < 16) {
    int z = z0 + tid; if (z >= npts) z = npts - 1;
    float px = r[3*z+0], py = r[3*z+1], pz = r[3*z+2];
    float rad = sqrtf(px*px + py*py + pz*pz);
    float inv = 1.0f / fmaxf(rad, 1e-12f);
    float x = px*inv, y = py*inv, zz = pz*inv;
    float x2=x*x, y2=y*y, z2=zz*zz;
    const float PI_F = 3.14159265358979323846f;
    float* Yp = sY + tid*32;
    Yp[0] = 0.5f*sqrtf(1.f/PI_F);
    float c1 = sqrtf(3.f/(4.f*PI_F));
    Yp[1]=c1*y; Yp[2]=c1*zz; Yp[3]=c1*x;
    float c15 = 0.5f*sqrtf(15.f/PI_F);
    Yp[4]=c15*x*y; Yp[5]=c15*y*zz;
    Yp[6]=0.25f*sqrtf(5.f/PI_F)*(3.f*z2-1.f);
    Yp[7]=c15*x*zz;
    Yp[8]=0.25f*sqrtf(15.f/PI_F)*(x2-y2);
    Yp[9] =0.25f*sqrtf(35.f/(2.f*PI_F))*y*(3.f*x2-y2);
    Yp[10]=0.5f *sqrtf(105.f/PI_F)*x*y*zz;
    Yp[11]=0.25f*sqrtf(21.f/(2.f*PI_F))*y*(5.f*z2-1.f);
    Yp[12]=0.25f*sqrtf(7.f/PI_F)*zz*(5.f*z2-3.f);
    Yp[13]=0.25f*sqrtf(21.f/(2.f*PI_F))*x*(5.f*z2-1.f);
    Yp[14]=0.25f*sqrtf(105.f/PI_F)*(x2-y2)*zz;
    Yp[15]=0.25f*sqrtf(35.f/(2.f*PI_F))*x*(x2-3.f*y2);
    Yp[16]=0.75f*sqrtf(35.f/PI_F)*x*y*(x2-y2);
    Yp[17]=0.75f*sqrtf(35.f/(2.f*PI_F))*y*(3.f*x2-y2)*zz;
    Yp[18]=0.75f*sqrtf(5.f/PI_F)*x*y*(7.f*z2-1.f);
    Yp[19]=0.75f*sqrtf(5.f/(2.f*PI_F))*y*zz*(7.f*z2-3.f);
    Yp[20]=(3.f/16.f)*sqrtf(1.f/PI_F)*(35.f*z2*z2-30.f*z2+3.f);
    Yp[21]=0.75f*sqrtf(5.f/(2.f*PI_F))*x*zz*(7.f*z2-3.f);
    Yp[22]=(3.f/8.f)*sqrtf(5.f/PI_F)*(x2-y2)*(7.f*z2-1.f);
    Yp[23]=0.75f*sqrtf(35.f/(2.f*PI_F))*x*(x2-3.f*y2)*zz;
    Yp[24]=(3.f/16.f)*sqrtf(35.f/PI_F)*(x2*x2-6.f*x2*y2+y2*y2);
    Yp[25]=rad;
    Yp[26]=(rad==0.f)?1.f:0.f;
  }
  __syncthreads();

  // ---- phase A2: h = relu(rad*W1+b1) -> f16 LDS tile (K padded) ----
  {
    int p   = tid & 15;
    int hh0 = tid >> 4;
    float rad = sY[p*32+25];
    for (int hh = hh0; hh < KPAD; hh += 16) {
      float v = 0.f;
      if (hh < HIDDEN) v = fmaxf(rad*W1[hh] + b1[hh], 0.f);
      sH[p*HSTRIDE + hh] = (_Float16)v;
    }
  }
  // ---- phase A3: CY[p][path][i,j] = sum_k C3j[i,j,k]*Y[k] ----
  computeCY< 0>(sCY,sY,sCG,tid); computeCY< 1>(sCY,sY,sCG,tid);
  computeCY< 2>(sCY,sY,sCG,tid); computeCY< 3>(sCY,sY,sCG,tid);
  computeCY< 4>(sCY,sY,sCG,tid); computeCY< 5>(sCY,sY,sCG,tid);
  computeCY< 6>(sCY,sY,sCG,tid); computeCY< 7>(sCY,sY,sCG,tid);
  computeCY< 8>(sCY,sY,sCG,tid); computeCY< 9>(sCY,sY,sCG,tid);
  computeCY<10>(sCY,sY,sCG,tid); computeCY<11>(sCY,sY,sCG,tid);
  computeCY<12>(sCY,sY,sCG,tid); computeCY<13>(sCY,sY,sCG,tid);
  computeCY<14>(sCY,sY,sCG,tid); computeCY<15>(sCY,sY,sCG,tid);
  computeCY<16>(sCY,sY,sCG,tid); computeCY<17>(sCY,sY,sCG,tid);
  computeCY<18>(sCY,sY,sCG,tid);
  __syncthreads();

  // ---- A fragments (contiguous k = 16h+e, same as B), reused by all tiles ----
  const int lane = tid & 31;
  const int am   = lane & 15;
  const int kb   = (lane & 16) ? 16 : 0;
  v16h a0 = loadA(sH, am, kb, 0);
  v16h a1 = loadA(sH, am, kb, 1);
  v16h a2 = loadA(sH, am, kb, 2);
  v16h a3 = loadA(sH, am, kb, 3);

  // ---- half 0: coeff cols [0,2560) -> output rows 0..63 ----
  gemmHalf<0,160>(sC, w2h, a0,a1,a2,a3, tid);
  __syncthreads();
  expandBlock<0,0,0>(sC,sCY,sY,sNorm,out,z0,npts,tid);
  expandBlock<0,1,0>(sC,sCY,sY,sNorm,out,z0,npts,tid);
  expandBlock<0,2,0>(sC,sCY,sY,sNorm,out,z0,npts,tid);
  expandBlock<1,0,0>(sC,sCY,sY,sNorm,out,z0,npts,tid);
  expandBlock<1,1,0>(sC,sCY,sY,sNorm,out,z0,npts,tid);
  expandBlock<1,2,0>(sC,sCY,sY,sNorm,out,z0,npts,tid);
  __syncthreads();

  // ---- half 1: coeff cols [2560,4864) -> output rows 64..143 ----
  gemmHalf<2560,144>(sC, w2h, a0,a1,a2,a3, tid);
  __syncthreads();
  expandBlock<2,0,2560>(sC,sCY,sY,sNorm,out,z0,npts,tid);
  expandBlock<2,1,2560>(sC,sCY,sY,sNorm,out,z0,npts,tid);
  expandBlock<2,2,2560>(sC,sCY,sY,sNorm,out,z0,npts,tid);
}

// =================================================================
// launcher
// =================================================================
extern "C" void kernel_launch(void* const* d_in, const int* in_sizes, int n_in,
                              void* d_out, int out_size, void* d_ws, size_t ws_size,
                              hipStream_t stream) {
  const float* r  = (const float*)d_in[0];
  const float* W1 = (const float*)d_in[1];
  const float* b1 = (const float*)d_in[2];
  const float* W2 = (const float*)d_in[3];
  float* out = (float*)d_out;

  char* ws = (char*)d_ws;
  float*    cg      = (float*)ws;                 // 1225 f32
  float*    normTab = (float*)(ws + 4928);        // 18 f32
  _Float16* w2h     = (_Float16*)(ws + 8192);     // 622,592 f16 (~1.2 MB)

  const int npts = in_sizes[0] / 3;               // 8192
  const int nwg  = (npts + 15) / 16;              // 512

  init_cg_kernel<<<1, 32, 0, stream>>>(cg, normTab);
  swizzle_w2<<<(NTILES*2048 + 255)/256, 256, 0, stream>>>(W2, w2h);

  (void)hipFuncSetAttribute((const void*)fused_kernel,
                            hipFuncAttributeMaxDynamicSharedMemorySize,
                            (int)SMEM_BYTES);
  fused_kernel<<<nwg, 256, SMEM_BYTES, stream>>>(r, W1, b1, cg, normTab, w2h, out, npts);
}
